// NewtonNet_13340168421978
// MI455X (gfx1250) — compile-verified, matrix operand order checked by reading, not run on hardware
//
#include <hip/hip_runtime.h>
#include <hip/hip_bf16.h>
#include <math.h>

#define F 128
#define NB 20

typedef __attribute__((ext_vector_type(16))) __bf16 v16bf;
typedef __attribute__((ext_vector_type(8)))  float  v8f;

struct BFrag { unsigned short u[16]; };   // 32 bytes
struct QFrag { uint4 a, b; };             // 32 bytes

__device__ __forceinline__ unsigned short f2bf(float f) {
  unsigned int x = __builtin_bit_cast(unsigned int, f);
  x = (x + 0x7FFFu + ((x >> 16) & 1u)) >> 16;
  return (unsigned short)x;
}

__device__ __forceinline__ void atomAddF(float* p, float v) {
  __hip_atomic_fetch_add(p, v, __ATOMIC_RELAXED, __HIP_MEMORY_SCOPE_AGENT);
}

// fast SiLU: x * rcp(1 + exp(-x))  -> v_exp_f32 + v_add + v_rcp_f32 + v_mul
__device__ __forceinline__ float siluf(float x) {
  return x * __builtin_amdgcn_rcpf(1.f + __expf(-x));
}

// ---------------------------------------------------------------------------
// Pack one 128x128 f32 weight (row-major [k][n]) into wave32 bf16 B-fragment
// layout: dst[((nt*4+kc)*32+lane)*16+e], n = nt*16+(lane&15),
// k = kc*32 + e + 8*(e>=8) + 8*(lane>=16).
// ---------------------------------------------------------------------------
__global__ __launch_bounds__(256) void k_pack(const float* __restrict__ W,
                                              unsigned short* __restrict__ dst) {
  int p = blockIdx.x * 256 + threadIdx.x;        // 64 blocks * 256 = 16384
  int e = p & 15, lane = (p >> 4) & 31, kc = (p >> 9) & 3, nt = (p >> 11) & 7;
  int n = nt * 16 + (lane & 15);
  int k = kc * 32 + e + ((e >= 8) ? 8 : 0) + ((lane >= 16) ? 8 : 0);
  dst[p] = f2bf(W[k * F + n]);
}

// ---------------------------------------------------------------------------
// Generic WMMA GEMM: out[r,c] = act( A[r,:] @ W + bias[c] ), K = N = 128.
// One wave = 16-row strip, 8 N-tiles x 4 K-steps = 32 v_wmma per wave.
// MODE 0: store (OUTB ? bf16 : f32).
// MODE 1: scatter em1 = val*disp_edge into f_agg via atomics.
// MODE 2: scatter em2 = val*disp[e1] into dacc via atomics.
// ---------------------------------------------------------------------------
template<int INBF, int SILU, int MODE, int OUTB>
__global__ __launch_bounds__(256) void k_gemm(
    const void* __restrict__ Ain, const unsigned short* __restrict__ Wpk,
    const float* __restrict__ bias, int nstrips,
    void* __restrict__ outp,
    const int* __restrict__ e0, const int* __restrict__ e1,
    const float* __restrict__ vec3, const float* __restrict__ gath,
    float* __restrict__ accb)
{
  const int lane  = threadIdx.x & 31;
  const int strip = blockIdx.x * 8 + (threadIdx.x >> 5);
  if (strip >= nstrips) return;
  const int lhi    = lane >> 4;                   // 0 or 1
  const int colb   = lane & 15;
  const int arow_i = strip * 16 + colb;           // A-fragment row
  const int rbase  = strip * 16 + (lhi << 3);     // C-fragment row base

  // Load the 4 A fragments covering K = 0..127.
  v16bf afrag[4];
  if (INBF) {
    const unsigned short* arow = (const unsigned short*)Ain + (size_t)arow_i * F;
    if (strip + 8 < nstrips) __builtin_prefetch(arow + 8 * 16 * F, 0, 1);
    #pragma unroll
    for (int kc = 0; kc < 4; ++kc) {
      int koff = kc * 32 + lhi * 8;
      QFrag q;
      q.a = *(const uint4*)(arow + koff);
      q.b = *(const uint4*)(arow + koff + 16);
      afrag[kc] = __builtin_bit_cast(v16bf, q);
    }
  } else {
    const float* arow = (const float*)Ain + (size_t)arow_i * F;
    if (strip + 8 < nstrips) __builtin_prefetch(arow + 8 * 16 * F, 0, 1);
    #pragma unroll
    for (int kc = 0; kc < 4; ++kc) {
      int koff = kc * 32 + lhi * 8;
      float fv[16];
      *(float4*)&fv[0]  = *(const float4*)(arow + koff);
      *(float4*)&fv[4]  = *(const float4*)(arow + koff + 4);
      *(float4*)&fv[8]  = *(const float4*)(arow + koff + 16);
      *(float4*)&fv[12] = *(const float4*)(arow + koff + 20);
      BFrag s;
      #pragma unroll
      for (int i = 0; i < 16; ++i) s.u[i] = f2bf(fv[i]);
      afrag[kc] = __builtin_bit_cast(v16bf, s);
    }
  }

  // Hoisted output bases: all per-(v,nt) offsets are compile-time immediates.
  float*          pf = (float*)outp          + (size_t)rbase * F + colb;
  unsigned short* pb = (unsigned short*)outp + (size_t)rbase * F + colb;
  unsigned bv[8], gv[8];                 // scatter bases (elements, < 2^31)
  float    d3[3][8];
  if (MODE == 1) {
    #pragma unroll
    for (int v = 0; v < 8; ++v) {
      int r = rbase + v;
      bv[v] = (unsigned)e0[r] * (3u * F) + colb;
      d3[0][v] = vec3[3 * r]; d3[1][v] = vec3[3 * r + 1]; d3[2][v] = vec3[3 * r + 2];
    }
  } else if (MODE == 2) {
    #pragma unroll
    for (int v = 0; v < 8; ++v) {
      int r = rbase + v;
      bv[v] = (unsigned)e0[r] * (3u * F) + colb;
      gv[v] = (unsigned)e1[r] * (3u * F) + colb;
    }
  }

  #pragma unroll
  for (int nt = 0; nt < 8; ++nt) {
    const float bval = bias ? bias[nt * 16 + colb] : 0.f;
    v8f acc;
    #pragma unroll
    for (int i = 0; i < 8; ++i) acc[i] = bval;    // bias folded into accumulator
    #pragma unroll
    for (int kc = 0; kc < 4; ++kc) {
      QFrag q = *(const QFrag*)(Wpk + ((size_t)(nt * 4 + kc) * 32 + lane) * 16);
      v16bf bf = __builtin_bit_cast(v16bf, q);
      acc = __builtin_amdgcn_wmma_f32_16x16x32_bf16(false, afrag[kc], false, bf,
                                                    (short)0, acc, false, false);
    }
    #pragma unroll
    for (int v = 0; v < 8; ++v) {
      float val = acc[v];
      if (SILU) val = siluf(val);
      if (MODE == 0) {
        if (OUTB) pb[v * F + nt * 16] = f2bf(val);
        else      pf[v * F + nt * 16] = val;
      } else if (MODE == 1) {
        float* base = accb + bv[v];
        #pragma unroll
        for (int d = 0; d < 3; ++d)
          atomAddF(base + d * F + nt * 16, val * d3[d][v]);
      } else if (MODE == 2) {
        float*       base = accb + bv[v];
        const float* gbase = gath + gv[v];
        #pragma unroll
        for (int d = 0; d < 3; ++d)
          atomAddF(base + d * F + nt * 16, val * gbase[d * F + nt * 16]);
      }
    }
  }
}

// ---------------------------------------------------------------------------
__global__ __launch_bounds__(256) void k_embed(const int* __restrict__ z,
                                               const float* __restrict__ emb,
                                               float* __restrict__ atom, int n) {
  int i = blockIdx.x * 256 + threadIdx.x;
  if (i < n * F) {
    int node = i >> 7, f = i & (F - 1);
    atom[i] = emb[(size_t)z[node] * F + f];
  }
}

__global__ __launch_bounds__(256) void k_zero(float* __restrict__ p, size_t n) {
  size_t i  = (size_t)blockIdx.x * 256 + threadIdx.x;
  size_t st = (size_t)gridDim.x * 256;
  for (; i < n; i += st) p[i] = 0.f;
}

__global__ __launch_bounds__(256) void k_add(float* __restrict__ out,
                                             const float* __restrict__ in, size_t n) {
  size_t i  = (size_t)blockIdx.x * 256 + threadIdx.x;
  size_t st = (size_t)gridDim.x * 256;
  for (; i < n; i += st) out[i] += in[i];
}

// disp_edge[e][3], dist_edge[e][20] = env * bessel rbf
__global__ __launch_bounds__(256) void k_edge_geom(const int* __restrict__ ei,
                                                   const float* __restrict__ pos,
                                                   const float* __restrict__ freq,
                                                   float* __restrict__ disp_edge,
                                                   float* __restrict__ dist_edge, int E) {
  int e = blockIdx.x * 256 + threadIdx.x;
  if (e >= E) return;
  int a = ei[e], b = ei[E + e];
  float dx = pos[a * 3]     - pos[b * 3];
  float dy = pos[a * 3 + 1] - pos[b * 3 + 1];
  float dz = pos[a * 3 + 2] - pos[b * 3 + 2];
  disp_edge[e * 3] = dx; disp_edge[e * 3 + 1] = dy; disp_edge[e * 3 + 2] = dz;
  float d = sqrtf(dx * dx + dy * dy + dz * dz + 1e-12f);
  float u = d * 0.2f;                               // d / RC, RC = 5
  float u2 = u * u, u4 = u2 * u2, u6 = u4 * u2, u7 = u6 * u, u8 = u4 * u4;
  float env = (d < 5.f) ? (1.f - 28.f * u6 + 48.f * u7 - 21.f * u8) : 0.f;
  float s = env * 0.6324555320336759f * __builtin_amdgcn_rcpf(d);  // sqrt(2/RC)/d
  #pragma unroll
  for (int k = 0; k < NB; ++k)
    dist_edge[(size_t)e * NB + k] = s * __sinf(d * freq[k]);
}

// inv_msg = (dist_edge @ We + be) * inv_node[e0] * inv_node[e1];
// atom[e0] += inv_msg (atomic segment-sum); store bf16 copy for eq GEMMs.
__global__ __launch_bounds__(128) void k_edge_msg(const int* __restrict__ ei,
                                                  const float* __restrict__ dist,
                                                  const float* __restrict__ We,
                                                  const float* __restrict__ be,
                                                  const float* __restrict__ inv_node,
                                                  float* __restrict__ atom,
                                                  unsigned short* __restrict__ msgb, int E) {
  int e = blockIdx.x, f = threadIdx.x;
  __shared__ float sd[NB];
  __shared__ int   se[2];
  if (f < NB) sd[f] = dist[(size_t)e * NB + f];
  if (f == 0) { se[0] = ei[e]; se[1] = ei[E + e]; }
  __syncthreads();
  float t = be[f];
  #pragma unroll
  for (int k = 0; k < NB; ++k) t = fmaf(sd[k], We[k * F + f], t);
  int n0 = se[0], n1 = se[1];
  float m = t * inv_node[(size_t)n0 * F + f] * inv_node[(size_t)n1 * F + f];
  atomAddF(&atom[(size_t)n0 * F + f], m);
  msgb[(size_t)e * F + f] = f2bf(m);
}

// dacc[e0] += eq3_inv[e1] * f_agg[e1]   (em3 segment-sum)
__global__ __launch_bounds__(128) void k_scatter3(const int* __restrict__ ei,
                                                  const float* __restrict__ eq3v,
                                                  const float* __restrict__ f_agg,
                                                  float* __restrict__ dacc, int E) {
  int e = blockIdx.x, f = threadIdx.x;
  int n0 = ei[e], n1 = ei[E + e];
  float s = eq3v[(size_t)n1 * F + f];
  const float* fb = f_agg + (size_t)n1 * 3 * F + f;
  float*       db = dacc  + (size_t)n0 * 3 * F + f;
  #pragma unroll
  for (int d = 0; d < 3; ++d)
    atomAddF(db + d * F, s * fb[d * F]);
}

// atom = LayerNorm(atom + updv * sum_d(-force*disp)); one wave32 per node
__global__ __launch_bounds__(256) void k_update(const float* __restrict__ updv,
                                                const float* __restrict__ forceb,
                                                const float* __restrict__ dispb,
                                                const float* __restrict__ lng,
                                                const float* __restrict__ lnb,
                                                float* __restrict__ atom, int n) {
  int node = blockIdx.x * 8 + (threadIdx.x >> 5);
  int lane = threadIdx.x & 31;
  if (node >= n) return;
  int fb = lane * 4;
  float4 a4 = *(const float4*)(atom + (size_t)node * F + fb);
  float4 u4 = *(const float4*)(updv + (size_t)node * F + fb);
  float dot[4] = {0.f, 0.f, 0.f, 0.f};
  #pragma unroll
  for (int d = 0; d < 3; ++d) {
    float4 fo = *(const float4*)(forceb + ((size_t)node * 3 + d) * F + fb);
    float4 di = *(const float4*)(dispb  + ((size_t)node * 3 + d) * F + fb);
    dot[0] -= fo.x * di.x; dot[1] -= fo.y * di.y;
    dot[2] -= fo.z * di.z; dot[3] -= fo.w * di.w;
  }
  float x[4] = { a4.x + u4.x * dot[0], a4.y + u4.y * dot[1],
                 a4.z + u4.z * dot[2], a4.w + u4.w * dot[3] };
  float s1 = x[0] + x[1] + x[2] + x[3];
  float s2 = x[0] * x[0] + x[1] * x[1] + x[2] * x[2] + x[3] * x[3];
  #pragma unroll
  for (int o = 16; o > 0; o >>= 1) {
    s1 += __shfl_xor(s1, o, 32);
    s2 += __shfl_xor(s2, o, 32);
  }
  float mu  = s1 * (1.f / 128.f);
  float var = s2 * (1.f / 128.f) - mu * mu;
  float inv = __builtin_amdgcn_rsqf(var + 1e-5f);
  float4 g4 = *(const float4*)(lng + fb);
  float4 b4 = *(const float4*)(lnb + fb);
  float4 r;
  r.x = (x[0] - mu) * inv * g4.x + b4.x;
  r.y = (x[1] - mu) * inv * g4.y + b4.y;
  r.z = (x[2] - mu) * inv * g4.z + b4.z;
  r.w = (x[3] - mu) * inv * g4.w + b4.w;
  *(float4*)(atom + (size_t)node * F + fb) = r;
}

// ---------------------------------------------------------------------------
extern "C" void kernel_launch(void* const* d_in, const int* in_sizes, int n_in,
                              void* d_out, int out_size, void* d_ws, size_t ws_size,
                              hipStream_t stream) {
  (void)in_sizes; (void)n_in; (void)out_size; (void)ws_size;
  const int N = 10000, E = 320000;

  const int*   z    = (const int*)d_in[0];
  const float* pos  = (const float*)d_in[1];
  const int*   ei   = (const int*)d_in[2];
  const float* emb  = (const float*)d_in[4];
  const float* freq = (const float*)d_in[5];

  float* atom  = (float*)d_out;
  float* force = atom  + (size_t)N * F;
  float* disp  = force + (size_t)N * 3 * F;

  char* ws = (char*)d_ws;
  size_t off = 0;
  auto take = [&](size_t bytes) -> char* {
    char* p = ws + off;
    off = (off + bytes + 255) & ~(size_t)255;
    return p;
  };
  unsigned short* wpk       = (unsigned short*)take((size_t)30 * 16384 * 2);
  float*          disp_edge = (float*)take((size_t)E * 3 * 4);
  float*          dist_edge = (float*)take((size_t)E * NB * 4);
  float*          inv_node  = (float*)take((size_t)N * F * 4);
  float*          eq3v      = (float*)take((size_t)N * F * 4);
  float*          updv      = (float*)take((size_t)N * F * 4);
  unsigned short* tmpN      = (unsigned short*)take((size_t)N * F * 2);
  float*          f_agg     = (float*)take((size_t)N * 3 * F * 4);
  float*          dacc      = (float*)take((size_t)N * 3 * F * 4);
  unsigned short* msgb      = (unsigned short*)take((size_t)E * F * 2);
  unsigned short* tmpE      = (unsigned short*)take((size_t)E * F * 2);

  // Init state + geometry
  k_embed<<<dim3((N * F + 255) / 256), 256, 0, stream>>>(z, emb, atom, N);
  k_zero<<<dim3(4096), 256, 0, stream>>>(force, (size_t)2 * N * 3 * F);
  k_edge_geom<<<dim3((E + 255) / 256), 256, 0, stream>>>(ei, pos, freq, disp_edge, dist_edge, E);

  // Pack all 30 dense weights (f32 -> bf16 WMMA B-fragment layout)
  for (int l = 0; l < 3; ++l) {
    int b = 6 + l * 22;
    const int widx[10] = { b + 0, b + 2, b + 6, b + 8, b + 10, b + 12, b + 14, b + 15, b + 16, b + 18 };
    for (int m = 0; m < 10; ++m)
      k_pack<<<dim3(64), 256, 0, stream>>>((const float*)d_in[widx[m]],
                                           wpk + (size_t)(l * 10 + m) * 16384);
  }

  const int sN = N / 16;             // 625 strips
  const int sE = E / 16;             // 20000 strips
  const int gN = (sN + 7) / 8;       // 79 blocks
  const int gE = (sE + 7) / 8;       // 2500 blocks

  for (int l = 0; l < 3; ++l) {
    int b = 6 + l * 22;
    const unsigned short* Wl = wpk + (size_t)l * 10 * 16384;
    const float* b_inv0 = (const float*)d_in[b + 1];
    const float* b_inv1 = (const float*)d_in[b + 3];
    const float* We     = (const float*)d_in[b + 4];
    const float* be     = (const float*)d_in[b + 5];
    const float* b_eq10 = (const float*)d_in[b + 7];
    const float* b_eq11 = (const float*)d_in[b + 9];
    const float* b_eq20 = (const float*)d_in[b + 11];
    const float* b_eq21 = (const float*)d_in[b + 13];
    const float* b_up0  = (const float*)d_in[b + 17];
    const float* b_up1  = (const float*)d_in[b + 19];
    const float* lng    = (const float*)d_in[b + 20];
    const float* lnb    = (const float*)d_in[b + 21];

    // inv_node = MLP(atom)
    k_gemm<0,1,0,1><<<gN,256,0,stream>>>(atom, Wl + 0*16384, b_inv0, sN, tmpN,
                                         nullptr, nullptr, nullptr, nullptr, nullptr);
    k_gemm<1,0,0,0><<<gN,256,0,stream>>>(tmpN, Wl + 1*16384, b_inv1, sN, inv_node,
                                         nullptr, nullptr, nullptr, nullptr, nullptr);
    k_zero<<<dim3(4096), 256, 0, stream>>>(f_agg, (size_t)N * 3 * F);
    k_zero<<<dim3(4096), 256, 0, stream>>>(dacc,  (size_t)N * 3 * F);

    // inv_msg + atom segment-sum
    k_edge_msg<<<dim3(E), 128, 0, stream>>>(ei, dist_edge, We, be, inv_node, atom, msgb, E);

    // eq1 MLP + fused em1 scatter into f_agg
    k_gemm<1,1,0,1><<<gE,256,0,stream>>>(msgb, Wl + 2*16384, b_eq10, sE, tmpE,
                                         nullptr, nullptr, nullptr, nullptr, nullptr);
    k_gemm<1,0,1,0><<<gE,256,0,stream>>>(tmpE, Wl + 3*16384, b_eq11, sE, nullptr,
                                         ei, ei + E, disp_edge, nullptr, f_agg);

    // eq2 MLP + fused em2 scatter (gather old disp[e1]) into dacc
    k_gemm<1,1,0,1><<<gE,256,0,stream>>>(msgb, Wl + 4*16384, b_eq20, sE, tmpE,
                                         nullptr, nullptr, nullptr, nullptr, nullptr);
    k_gemm<1,0,2,0><<<gE,256,0,stream>>>(tmpE, Wl + 5*16384, b_eq21, sE, nullptr,
                                         ei, ei + E, nullptr, disp, dacc);

    // force += f_agg
    k_add<<<dim3(4096), 256, 0, stream>>>(force, f_agg, (size_t)N * 3 * F);

    // eq3 (no-bias MLP on updated atom)
    k_gemm<0,1,0,1><<<gN,256,0,stream>>>(atom, Wl + 6*16384, nullptr, sN, tmpN,
                                         nullptr, nullptr, nullptr, nullptr, nullptr);
    k_gemm<1,0,0,0><<<gN,256,0,stream>>>(tmpN, Wl + 7*16384, nullptr, sN, eq3v,
                                         nullptr, nullptr, nullptr, nullptr, nullptr);
    k_scatter3<<<dim3(E), 128, 0, stream>>>(ei, eq3v, f_agg, dacc, E);

    // disp += dacc (em2 + em3)
    k_add<<<dim3(4096), 256, 0, stream>>>(disp, dacc, (size_t)N * 3 * F);

    // upd MLP + gated residual + LayerNorm
    k_gemm<0,1,0,1><<<gN,256,0,stream>>>(atom, Wl + 8*16384, b_up0, sN, tmpN,
                                         nullptr, nullptr, nullptr, nullptr, nullptr);
    k_gemm<1,0,0,0><<<gN,256,0,stream>>>(tmpN, Wl + 9*16384, b_up1, sN, updv,
                                         nullptr, nullptr, nullptr, nullptr, nullptr);
    k_update<<<dim3((N + 7) / 8), 256, 0, stream>>>(updv, force, disp, lng, lnb, atom, N);
  }
}
